// GCN_53472342835548
// MI455X (gfx1250) — matrix operand order repaired and checked by reading, not compiled, not run on hardware
//
#include <hip/hip_runtime.h>
#include <math.h>

typedef __attribute__((ext_vector_type(16))) _Float16 v16h;
typedef __attribute__((ext_vector_type(2)))  _Float16 h2v;
typedef __attribute__((ext_vector_type(8)))  float    v8f;

#define F_IN 128
#define HDIM 64
#define CDIM 40
#define KEEP 0.5856f

__device__ __forceinline__ void atomicAddF(float* p, float v) {
    // lowers to global_atomic_add_f32 (native), not a CAS loop
    __hip_atomic_fetch_add(p, v, __ATOMIC_RELAXED, __HIP_MEMORY_SCOPE_AGENT);
}

__device__ __forceinline__ float hash01(unsigned int x) {
    x ^= 42u * 0x9E3779B9u;
    x ^= x >> 16; x *= 0x7feb352du;
    x ^= x >> 15; x *= 0x846ca68bu;
    x ^= x >> 16;
    return (float)(x >> 8) * (1.0f / 16777216.0f);
}

// Fragment-order LDS indexing helpers (16-bit A 16x32, contiguous k order k = 16*hf + s):
//   within a 32-wide K step, element (m, kw) lives at lane = hf*16+m,
//   slot s, where hf=kw>>4 and s=kw&15.
// B (32x16): element (kw, n) lives at lane = (kw>>4)*16+n, slot kw&15.

// ---------------- degree / normalization ----------------
__global__ void deg_init_kernel(float* deg, int N) {
    int i = blockIdx.x * blockDim.x + threadIdx.x;
    if (i < N) deg[i] = 1.0f;   // self-loop
}

__global__ void deg_acc_kernel(const int* __restrict__ col, float* deg, int E) {
    int e = blockIdx.x * blockDim.x + threadIdx.x;
    if (e < E) atomicAddF(&deg[col[e]], 1.0f);
}

__global__ void deg_rsqrt_kernel(float* deg, int N) {
    int i = blockIdx.x * blockDim.x + threadIdx.x;
    if (i < N) deg[i] = rsqrtf(fmaxf(deg[i], 1.0f));
}

// ---------------- GEMM1: h1[N,64] = x[N,128] @ W1[128,64], f16 WMMA ----------------
__global__ void gemm1_kernel(const float* __restrict__ x, const float* __restrict__ W1,
                             float* __restrict__ h1, int N) {
    // fragment-order tiles: [wavetile|kstep][lane][16 halfs]
    __shared__ _Float16 AsF[8 * 4 * 32 * 16];   // 32 KB
    __shared__ _Float16 BsF[4 * 4 * 32 * 16];   // 16 KB
    int tid = threadIdx.x;                      // 256 threads = 8 waves
    int rowBase = blockIdx.x * 128;

    // stage W1 (128x64) into B fragment order
    for (int q = tid; q < F_IN * HDIM; q += 256) {
        int k = q >> 6, c = q & 63;
        int ks = k >> 5, kw = k & 31;
        int hf = kw >> 4, s = kw & 15;
        int nt = c >> 4, n = c & 15;
        BsF[(((ks * 4 + nt) * 32) + hf * 16 + n) * 16 + s] = (_Float16)W1[q];
    }
    // stage x tile (128x128) into A fragment order, pair granularity
    for (int q = tid; q < 128 * 64; q += 256) {
        int r = q >> 6, pw = q & 63;
        int kglob = pw << 1;
        int gr = rowBase + r;
        float v0 = 0.0f, v1 = 0.0f;
        if (gr < N) {
            float2 xv = *(const float2*)&x[(size_t)gr * F_IN + kglob];
            v0 = xv.x; v1 = xv.y;
        }
        int ks = kglob >> 5, kw = kglob & 31;
        int hf = kw >> 4;
        int s  = kw & 15;   // even
        int wt = r >> 4, m = r & 15;
        h2v pk = { (_Float16)v0, (_Float16)v1 };
        *(h2v*)&AsF[(((wt * 4 + ks) * 32) + hf * 16 + m) * 16 + s] = pk;
    }
    __syncthreads();

    int wave = tid >> 5, lane = tid & 31;
    if (rowBase + wave * 16 >= N) return;       // wave-uniform: EXEC all-1s for WMMA
    const _Float16* aBase = &AsF[(wave * 4 * 32 + lane) * 16];
    const _Float16* bBase = &BsF[lane * 16];

    v8f acc[4] = {};
#pragma unroll
    for (int ks = 0; ks < 4; ++ks) {
        v16h a = *(const v16h*)(aBase + ks * 32 * 16);
#pragma unroll
        for (int nt = 0; nt < 4; ++nt) {
            v16h b = *(const v16h*)(bBase + (ks * 4 + nt) * 32 * 16);
            acc[nt] = __builtin_amdgcn_wmma_f32_16x16x32_f16(
                false, a, false, b, (short)0, acc[nt], false, false);
        }
    }
    int hf = lane >> 4, n = lane & 15;
    int gr0 = rowBase + wave * 16;
#pragma unroll
    for (int nt = 0; nt < 4; ++nt)
#pragma unroll
        for (int r = 0; r < 8; ++r)
            h1[(size_t)(gr0 + r + 8 * hf) * HDIM + nt * 16 + n] = acc[nt][r];
}

// ---------------- edge aggregation: out[col] += h[row] * dinv[row]*dinv[col] ----------------
__global__ void agg_kernel(const float* __restrict__ h, const int* __restrict__ rowi,
                           const int* __restrict__ coli, const float* __restrict__ dinv,
                           float* __restrict__ out, int E, int F) {
    int wid  = (blockIdx.x * blockDim.x + threadIdx.x) >> 5;
    int lane = threadIdx.x & 31;
    int nw   = (gridDim.x * blockDim.x) >> 5;
    for (int e = wid; e < E; e += nw) {
        int r = rowi[e], c = coli[e];
        float nrm = dinv[r] * dinv[c];
        size_t rb = (size_t)r * F, cb = (size_t)c * F;
        for (int f = lane; f < F; f += 32)
            atomicAddF(&out[cb + f], h[rb + f] * nrm);
    }
}

// ---------------- bias + relu + dropout (writes d1 into h1 buffer) ----------------
__global__ void post1_kernel(const float* __restrict__ agg1, float* __restrict__ h1,
                             const float* __restrict__ dinv, const float* __restrict__ b1,
                             int N) {
    int i = blockIdx.x * blockDim.x + threadIdx.x;
    int total = N * HDIM;
    if (i >= total) return;
    int node = i >> 6, f = i & 63;
    float di = dinv[node];
    float v = agg1[i] + h1[i] * di * di + b1[f];
    v = fmaxf(v, 0.0f);
    float u = hash01((unsigned int)i);
    v = (u < KEEP) ? v * (1.0f / KEEP) : 0.0f;
    h1[i] = v;
}

// ---------------- GEMM2: h2[N,40] = d1[N,64] @ W2[64,40], f16 WMMA (pad 40->48) ----------------
__global__ void gemm2_kernel(const float* __restrict__ d1, const float* __restrict__ W2,
                             float* __restrict__ h2, int N) {
    __shared__ _Float16 AsF[8 * 2 * 32 * 16];   // 16 KB
    __shared__ _Float16 BsF[2 * 3 * 32 * 16];   // 6 KB
    int tid = threadIdx.x;
    int rowBase = blockIdx.x * 128;

    // stage W2 (64x40 -> 64x48 padded) into B fragment order
    for (int q = tid; q < HDIM * 48; q += 256) {
        int k = q / 48, c = q % 48;
        int ks = k >> 5, kw = k & 31;
        int hf = kw >> 4, s = kw & 15;
        int nt = c >> 4, n = c & 15;
        _Float16 v = (c < CDIM) ? (_Float16)W2[k * CDIM + c] : (_Float16)0.0f;
        BsF[(((ks * 3 + nt) * 32) + hf * 16 + n) * 16 + s] = v;
    }
    // stage d1 tile (128x64) into A fragment order, pair granularity
    for (int q = tid; q < 128 * 32; q += 256) {
        int r = q >> 5, pw = q & 31;
        int kglob = pw << 1;
        int gr = rowBase + r;
        float v0 = 0.0f, v1 = 0.0f;
        if (gr < N) {
            float2 xv = *(const float2*)&d1[(size_t)gr * HDIM + kglob];
            v0 = xv.x; v1 = xv.y;
        }
        int ks = kglob >> 5, kw = kglob & 31;
        int hf = kw >> 4;
        int s  = kw & 15;
        int wt = r >> 4, m = r & 15;
        h2v pk = { (_Float16)v0, (_Float16)v1 };
        *(h2v*)&AsF[(((wt * 2 + ks) * 32) + hf * 16 + m) * 16 + s] = pk;
    }
    __syncthreads();

    int wave = tid >> 5, lane = tid & 31;
    if (rowBase + wave * 16 >= N) return;
    const _Float16* aBase = &AsF[(wave * 2 * 32 + lane) * 16];
    const _Float16* bBase = &BsF[lane * 16];

    v8f acc[3] = {};
#pragma unroll
    for (int ks = 0; ks < 2; ++ks) {
        v16h a = *(const v16h*)(aBase + ks * 32 * 16);
#pragma unroll
        for (int nt = 0; nt < 3; ++nt) {
            v16h b = *(const v16h*)(bBase + (ks * 3 + nt) * 32 * 16);
            acc[nt] = __builtin_amdgcn_wmma_f32_16x16x32_f16(
                false, a, false, b, (short)0, acc[nt], false, false);
        }
    }
    int hf = lane >> 4, n = lane & 15;
    int gr0 = rowBase + wave * 16;
#pragma unroll
    for (int nt = 0; nt < 3; ++nt)
#pragma unroll
        for (int r = 0; r < 8; ++r) {
            int c = nt * 16 + n;
            if (c < CDIM)
                h2[(size_t)(gr0 + r + 8 * hf) * CDIM + c] = acc[nt][r];
        }
}

// ---------------- self term + bias + log_softmax (wave per row) ----------------
__global__ void final_lsm_kernel(float* __restrict__ out, const float* __restrict__ h2,
                                 const float* __restrict__ dinv, const float* __restrict__ b2,
                                 int N) {
    int wid  = (blockIdx.x * blockDim.x + threadIdx.x) >> 5;
    int lane = threadIdx.x & 31;
    if (wid >= N) return;
    float di = dinv[wid];
    float di2 = di * di;
    size_t base = (size_t)wid * CDIM;
    float p0 = out[base + lane] + h2[base + lane] * di2 + b2[lane];
    bool has1 = lane < (CDIM - 32);
    float p1 = has1 ? (out[base + lane + 32] + h2[base + lane + 32] * di2 + b2[lane + 32])
                    : -INFINITY;
    float mx = fmaxf(p0, p1);
#pragma unroll
    for (int off = 16; off > 0; off >>= 1) mx = fmaxf(mx, __shfl_xor(mx, off, 32));
    float s = expf(p0 - mx) + (has1 ? expf(p1 - mx) : 0.0f);
#pragma unroll
    for (int off = 16; off > 0; off >>= 1) s += __shfl_xor(s, off, 32);
    float ls = logf(s);
    out[base + lane] = p0 - mx - ls;
    if (has1) out[base + lane + 32] = p1 - mx - ls;
}

extern "C" void kernel_launch(void* const* d_in, const int* in_sizes, int n_in,
                              void* d_out, int out_size, void* d_ws, size_t ws_size,
                              hipStream_t stream) {
    const float* x   = (const float*)d_in[0];
    const int*   ei  = (const int*)d_in[1];
    const float* W1  = (const float*)d_in[2];
    const float* b1  = (const float*)d_in[3];
    const float* W2  = (const float*)d_in[4];
    const float* b2  = (const float*)d_in[5];

    const int N = in_sizes[0] / F_IN;
    const int E = in_sizes[1] / 2;
    const int* rowi = ei;       // edge_index[0]
    const int* coli = ei + E;   // edge_index[1]

    float* deg  = (float*)d_ws;                 // N  (becomes dinv)
    float* h1   = deg + N;                      // N*64 (becomes d1)
    float* agg1 = h1 + (size_t)N * HDIM;        // N*64
    float* h2   = agg1 + (size_t)N * HDIM;      // N*40
    float* outp = (float*)d_out;                // N*40 accumulator -> final

    hipMemsetAsync(agg1, 0, (size_t)N * HDIM * sizeof(float), stream);
    hipMemsetAsync(outp, 0, (size_t)N * CDIM * sizeof(float), stream);

    // degree + dinv
    deg_init_kernel<<<(N + 255) / 256, 256, 0, stream>>>(deg, N);
    deg_acc_kernel<<<(E + 255) / 256, 256, 0, stream>>>(coli, deg, E);
    deg_rsqrt_kernel<<<(N + 255) / 256, 256, 0, stream>>>(deg, N);

    // layer 1
    gemm1_kernel<<<(N + 127) / 128, 256, 0, stream>>>(x, W1, h1, N);
    {
        int blocks = (E + 7) / 8;               // one wave per edge (grid-stride)
        if (blocks > 131072) blocks = 131072;
        agg_kernel<<<blocks, 256, 0, stream>>>(h1, rowi, coli, deg, agg1, E, HDIM);
    }
    post1_kernel<<<(N * HDIM + 255) / 256, 256, 0, stream>>>(agg1, h1, deg, b1, N);

    // layer 2
    gemm2_kernel<<<(N + 127) / 128, 256, 0, stream>>>(h1, W2, h2, N);
    {
        int blocks = (E + 7) / 8;
        if (blocks > 131072) blocks = 131072;
        agg_kernel<<<blocks, 256, 0, stream>>>(h2, rowi, coli, deg, outp, E, CDIM);
    }
    final_lsm_kernel<<<(N + 7) / 8, 256, 0, stream>>>(outp, h2, deg, b2, N);
}
